// LNLinearSigmoid_12833362280974
// MI455X (gfx1250) — matrix operand order repaired and checked by reading, not compiled, not run on hardware
//
#include <hip/hip_runtime.h>

// ---------------------------------------------------------------------------
// Types for CDNA5 WMMA
// ---------------------------------------------------------------------------
typedef __attribute__((ext_vector_type(16))) __bf16 v16bf;
typedef __attribute__((ext_vector_type(8)))  float  v8f;

union BFrag {
    uint4 q[2];   // 32 bytes = 16 bf16
    v16bf v;
};

// Pointer types for the async-to-LDS builtin (per hipcc diagnostic: it takes
// pointers to a 16-byte int vector; global side is AS(1), LDS side AS(3)).
typedef int v4i __attribute__((vector_size(16)));
typedef __attribute__((address_space(1))) v4i gv4i;
typedef __attribute__((address_space(3))) v4i lv4i;

#if defined(__has_builtin)
#if __has_builtin(__builtin_amdgcn_global_load_async_to_lds_b128)
#define HAVE_ASYNC_LDS 1
#pragma message("CDNA5: using __builtin_amdgcn_global_load_async_to_lds_b128")
#endif
#endif
#ifndef HAVE_ASYNC_LDS
#define HAVE_ASYNC_LDS 0
#pragma message("CDNA5: async-to-LDS builtin NOT available, using sync copy fallback")
#endif

__device__ __forceinline__ void copy16_to_lds(const unsigned short* g, unsigned short* s) {
#if HAVE_ASYNC_LDS
    __builtin_amdgcn_global_load_async_to_lds_b128(
        (gv4i*)g, (lv4i*)s, /*offset=*/0, /*cpol=*/0);
#else
    *(uint4*)s = *(const uint4*)g;
#endif
}

__device__ __forceinline__ void wait_async_copies() {
#if HAVE_ASYNC_LDS
#if defined(__has_builtin) && __has_builtin(__builtin_amdgcn_s_wait_asynccnt)
    __builtin_amdgcn_s_wait_asynccnt(0);
#else
    asm volatile("s_wait_asynccnt 0x0" ::: "memory");
#endif
#endif
}

// Round-to-nearest-even fp32 -> bf16 (bit pattern in ushort)
__device__ __forceinline__ unsigned short f2bf(float f) {
    unsigned int u = __float_as_uint(f);
    u += 0x7FFFu + ((u >> 16) & 1u);
    return (unsigned short)(u >> 16);
}

__device__ __forceinline__ unsigned int pack2bf(float lo, float hi) {
    return (unsigned int)f2bf(lo) | ((unsigned int)f2bf(hi) << 16);
}

// ---------------------------------------------------------------------------
// Kernel 1: LayerNorm (no affine, eps=1e-5) fp32 -> bf16
// One block (256 threads) per row of 2048.
// ---------------------------------------------------------------------------
__global__ __launch_bounds__(256) void ln_bf16_kernel(
    const float* __restrict__ x, unsigned short* __restrict__ xn)
{
    const int COLS = 2048;
    const int row  = blockIdx.x;
    const int tid  = threadIdx.x;

    const float4* xr = (const float4*)(x + (size_t)row * COLS);
    float4 a = xr[tid * 2 + 0];
    float4 b = xr[tid * 2 + 1];

    float s  = a.x + a.y + a.z + a.w + b.x + b.y + b.z + b.w;
    float s2 = a.x*a.x + a.y*a.y + a.z*a.z + a.w*a.w
             + b.x*b.x + b.y*b.y + b.z*b.z + b.w*b.w;

    __shared__ float r_s[256];
    __shared__ float r_q[256];
    r_s[tid] = s; r_q[tid] = s2;
    __syncthreads();
    for (int off = 128; off > 0; off >>= 1) {
        if (tid < off) { r_s[tid] += r_s[tid + off]; r_q[tid] += r_q[tid + off]; }
        __syncthreads();
    }
    const float inv_n = 1.0f / (float)COLS;
    float mean = r_s[0] * inv_n;
    float var  = r_q[0] * inv_n - mean * mean;
    float rstd = rsqrtf(var + 1e-5f);

    float v0 = (a.x - mean) * rstd, v1 = (a.y - mean) * rstd;
    float v2 = (a.z - mean) * rstd, v3 = (a.w - mean) * rstd;
    float v4 = (b.x - mean) * rstd, v5 = (b.y - mean) * rstd;
    float v6 = (b.z - mean) * rstd, v7 = (b.w - mean) * rstd;

    uint4 o;
    o.x = pack2bf(v0, v1); o.y = pack2bf(v2, v3);
    o.z = pack2bf(v4, v5); o.w = pack2bf(v6, v7);
    ((uint4*)(xn + (size_t)row * COLS))[tid] = o;
}

// ---------------------------------------------------------------------------
// Kernel 2: W fp32 -> bf16 (8 elements / thread)
// ---------------------------------------------------------------------------
__global__ __launch_bounds__(256) void wconv_kernel(
    const float* __restrict__ W, unsigned short* __restrict__ Wb)
{
    const size_t idx = (size_t)blockIdx.x * 256 + threadIdx.x;   // handles 8 elems
    const float4* p = (const float4*)W + idx * 2;
    float4 a = p[0];
    float4 b = p[1];
    uint4 o;
    o.x = pack2bf(a.x, a.y); o.y = pack2bf(a.z, a.w);
    o.z = pack2bf(b.x, b.y); o.w = pack2bf(b.z, b.w);
    ((uint4*)Wb)[idx] = o;
}

// ---------------------------------------------------------------------------
// Kernel 3: y = sigmoid(xn @ W^T), bf16 WMMA with async-LDS double buffering
//   A = xn [M=16384, K=2048] bf16 row-major
//   B = Wb [N=2048,  K=2048] bf16 row-major  (B_kxn[k][n] = Wb[n][k])
// Block: 256 threads = 8 waves in 4(M) x 2(N) grid -> 128 x 128 block tile.
// Wave tile: 32 x 64 = 2 x 4 accumulators of 16x16.
// K-step 32: A tile 128x32, B tile 128x32 staged in LDS (padded stride 40).
// ---------------------------------------------------------------------------
#define GK 2048
#define GN 2048
#define BM 128
#define BN 128
#define BK 32
#define LSTRIDE 40          // bf16 elems per LDS row (80 B, bank-conflict-free)

__global__ __launch_bounds__(256) void gemm_bf16_sigmoid_kernel(
    const unsigned short* __restrict__ A,
    const unsigned short* __restrict__ B,
    float* __restrict__ out)
{
    __shared__ __align__(16) unsigned short shA[2][BM * LSTRIDE];  // 2 x 10 KB
    __shared__ __align__(16) unsigned short shB[2][BN * LSTRIDE];  // 2 x 10 KB

    const int tid   = threadIdx.x;
    const int wave  = tid >> 5;
    const int lane  = tid & 31;
    const int half  = lane >> 4;     // 0: lanes 0-15, 1: lanes 16-31
    const int lr    = lane & 15;

    const int waveM = wave >> 1;     // 0..3
    const int waveN = wave & 1;      // 0..1
    const int wm    = waveM * 32;    // wave row offset inside block tile
    const int wn    = waveN * 64;    // wave col offset inside block tile

    const int blockM = blockIdx.x * BM;
    const int blockN = blockIdx.y * BN;

    // --- staging indices: thread copies 16B segments of rows crow, crow+64 ---
    const int crow = tid >> 2;       // 0..63
    const int cseg = tid & 3;        // 0..3 (16B segment within 64B row)
    const unsigned short* gA0 = A + (size_t)(blockM + crow)      * GK + cseg * 8;
    const unsigned short* gA1 = A + (size_t)(blockM + crow + 64) * GK + cseg * 8;
    const unsigned short* gB0 = B + (size_t)(blockN + crow)      * GK + cseg * 8;
    const unsigned short* gB1 = B + (size_t)(blockN + crow + 64) * GK + cseg * 8;
    const int sOff0 = crow        * LSTRIDE + cseg * 8;
    const int sOff1 = (crow + 64) * LSTRIDE + cseg * 8;

    v8f acc[2][4];
#pragma unroll
    for (int mi = 0; mi < 2; ++mi)
#pragma unroll
        for (int ni = 0; ni < 4; ++ni)
            acc[mi][ni] = (v8f){0.f,0.f,0.f,0.f,0.f,0.f,0.f,0.f};

    // Prologue: stage k=0 into buffer 0.
    copy16_to_lds(gA0, &shA[0][sOff0]);
    copy16_to_lds(gA1, &shA[0][sOff1]);
    copy16_to_lds(gB0, &shB[0][sOff0]);
    copy16_to_lds(gB1, &shB[0][sOff1]);

    int buf = 0;
    for (int k = 0; k < GK; k += BK) {
        wait_async_copies();
        __syncthreads();                       // tiles[buf] ready; prior reads done

        if (k + BK < GK) {                     // stage next K-step into other buffer
            const int nb = buf ^ 1;
            copy16_to_lds(gA0 + k + BK, &shA[nb][sOff0]);
            copy16_to_lds(gA1 + k + BK, &shA[nb][sOff1]);
            copy16_to_lds(gB0 + k + BK, &shB[nb][sOff0]);
            copy16_to_lds(gB1 + k + BK, &shB[nb][sOff1]);
        }

        // --- fragment loads from LDS (ds_load_b128) ---
        // A frag (16-bit 16x32): lane row = wm + mi*16 + lr
        //   V0..V3: K = 8*half + {0..7}  (16B); V4..V7: +16 elems (+32B)
        BFrag af[2], bfr[4];
#pragma unroll
        for (int mi = 0; mi < 2; ++mi) {
            const unsigned short* p = &shA[buf][(wm + mi * 16 + lr) * LSTRIDE + half * 8];
            af[mi].q[0] = *(const uint4*)p;
            af[mi].q[1] = *(const uint4*)(p + 16);
        }
        // B frag (16-bit 32x16): lane col = wn + ni*16 + lr
        //   V0..V3: K = 8*half + {0..7}  (16B); V4..V7: +16 elems (+32B)
#pragma unroll
        for (int ni = 0; ni < 4; ++ni) {
            const unsigned short* p = &shB[buf][(wn + ni * 16 + lr) * LSTRIDE + half * 8];
            bfr[ni].q[0] = *(const uint4*)p;
            bfr[ni].q[1] = *(const uint4*)(p + 16);
        }

#pragma unroll
        for (int mi = 0; mi < 2; ++mi)
#pragma unroll
            for (int ni = 0; ni < 4; ++ni)
                acc[mi][ni] = __builtin_amdgcn_wmma_f32_16x16x32_bf16(
                    /*neg_a=*/false, af[mi].v,
                    /*neg_b=*/false, bfr[ni].v,
                    /*c_mod=*/(short)0, acc[mi][ni],
                    /*reuse_a=*/false, /*reuse_b=*/false);

        buf ^= 1;
    }

    // Epilogue: sigmoid + store f32.
    // C/D layout: VGPR v -> row = v + 8*half of the tile, lane lr -> col.
#pragma unroll
    for (int mi = 0; mi < 2; ++mi) {
#pragma unroll
        for (int ni = 0; ni < 4; ++ni) {
#pragma unroll
            for (int v = 0; v < 8; ++v) {
                int row = blockM + wm + mi * 16 + v + 8 * half;
                int col = blockN + wn + ni * 16 + lr;
                float y = acc[mi][ni][v];
                out[(size_t)row * GN + col] = 1.0f / (1.0f + __expf(-y));
            }
        }
    }
}

// ---------------------------------------------------------------------------
// Launch
// ---------------------------------------------------------------------------
extern "C" void kernel_launch(void* const* d_in, const int* in_sizes, int n_in,
                              void* d_out, int out_size, void* d_ws, size_t ws_size,
                              hipStream_t stream)
{
    const float* x = (const float*)d_in[0];   // [16384, 2048]
    const float* W = (const float*)d_in[1];   // [2048, 2048]
    float* out     = (float*)d_out;           // [16384, 2048]

    unsigned short* xn = (unsigned short*)d_ws;                       // 64 MB
    unsigned short* Wb = xn + (size_t)16384 * 2048;                   // +8 MB

    ln_bf16_kernel<<<16384, 256, 0, stream>>>(x, xn);
    wconv_kernel<<<(2048 * 2048) / (256 * 8), 256, 0, stream>>>(W, Wb);

    dim3 grid(16384 / BM, 2048 / BN);   // (128, 16)
    gemm_bf16_sigmoid_kernel<<<grid, 256, 0, stream>>>(xn, Wb, out);
}
